// ProximalLSTMCell_29626684408120
// MI455X (gfx1250) — compile-verified, matrix operand order checked by reading, not run on hardware
//
#include <hip/hip_runtime.h>
#include <math.h>
#include <stdint.h>

typedef __attribute__((ext_vector_type(2)))  float  v2f;
typedef __attribute__((ext_vector_type(8)))  float  v8f;
typedef __attribute__((ext_vector_type(8)))  __bf16 v8bf;
typedef __attribute__((ext_vector_type(16))) __bf16 v16bf;

#define EPSV 1.0f
#define NB   256   // batch
#define HH   512   // hidden
#define DD   512   // input
#define CG_ITERS 64
#define KCH  32                    // K chunk per WMMA step
#define KBIG (3*HH)                // 1536
#define NCHUNK (KBIG / KCH)        // 48

// ---------------------------------------------------------------------------
// f32 TN GEMM (exact):  C[m,p] (+)= sum_k A[m,k]*B[p,k]
// Used for the one-time GEMMs (z pre-activations, weight Gram) where full
// fp32 precision sets the accuracy floor.  16(M) x 64(N) tile per wave.
// ---------------------------------------------------------------------------
__global__ __launch_bounds__(256) void gemm_tn_wmma(
    float* __restrict__ C, const float* __restrict__ A, const float* __restrict__ B,
    int M, int N, int K, int lda, int ldb, int ldc, int accumulate)
{
    const int lane   = threadIdx.x & 31;
    const int waveId = (blockIdx.x << 3) + (threadIdx.x >> 5);
    const int tilesM = M >> 4;
    const int tilesN = N >> 6;
    if (waveId >= tilesM * tilesN) return;          // uniform per wave: EXEC stays full
    const int m0 = (waveId % tilesM) << 4;
    const int p0 = (waveId / tilesM) << 6;

    const int khalf = lane >> 4;        // 0 -> K+0..1, 1 -> K+2..3
    const int row   = lane & 15;

    v8f acc0 = {}, acc1 = {}, acc2 = {}, acc3 = {};
    if (accumulate) {
#pragma unroll
        for (int v = 0; v < 8; ++v) {
            const int r = m0 + v + khalf * 8;
            const size_t cb = (size_t)r * ldc + p0 + row;
            acc0[v] = C[cb];      acc1[v] = C[cb + 16];
            acc2[v] = C[cb + 32]; acc3[v] = C[cb + 48];
        }
    }

    const float* aRow = A + (size_t)(m0 + row) * lda + 2 * khalf;
    const float* b0   = B + (size_t)(p0 + row) * ldb + 2 * khalf;
    const float* b1   = b0 + (size_t)16 * ldb;
    const float* b2   = b0 + (size_t)32 * ldb;
    const float* b3   = b0 + (size_t)48 * ldb;

    for (int k = 0; k < K; k += 4) {
        v2f a   = *(const v2f*)(aRow + k);
        v2f bb0 = *(const v2f*)(b0 + k);
        v2f bb1 = *(const v2f*)(b1 + k);
        v2f bb2 = *(const v2f*)(b2 + k);
        v2f bb3 = *(const v2f*)(b3 + k);
        acc0 = __builtin_amdgcn_wmma_f32_16x16x4_f32(false, a, false, bb0, (short)0, acc0, false, false);
        acc1 = __builtin_amdgcn_wmma_f32_16x16x4_f32(false, a, false, bb1, (short)0, acc1, false, false);
        acc2 = __builtin_amdgcn_wmma_f32_16x16x4_f32(false, a, false, bb2, (short)0, acc2, false, false);
        acc3 = __builtin_amdgcn_wmma_f32_16x16x4_f32(false, a, false, bb3, (short)0, acc3, false, false);
    }

#pragma unroll
    for (int v = 0; v < 8; ++v) {
        const int r = m0 + v + khalf * 8;
        const size_t cb = (size_t)r * ldc + p0 + row;
        C[cb]      = acc0[v];
        C[cb + 16] = acc1[v];
        C[cb + 32] = acc2[v];
        C[cb + 48] = acc3[v];
    }
}

// ---------------------------------------------------------------------------
// bf16 16x32 fragment load (TN operand, K contiguous in a 32-elem row slice).
// ISA layout: lanes 0-15 hold K=[k..k+7] in V0-3 and K=[k+16..k+23] in V4-7;
// lanes 16-31 hold K=[k+8..k+15] and K=[k+24..k+31]  ->  base = k + 8*khalf.
// Works for both global and LDS pointers (compiler picks vmem vs ds loads).
// ---------------------------------------------------------------------------
__device__ __forceinline__ v16bf load_frag_bf16(const __bf16* p) {
    v8bf lo = *(const v8bf*)p;          // 16 bytes: K base .. base+7
    v8bf hi = *(const v8bf*)(p + 16);   // 16 bytes: K base+16 .. base+23
    return __builtin_shufflevector(lo, hi, 0,1,2,3,4,5,6,7,8,9,10,11,12,13,14,15);
}

__device__ __forceinline__ void async_b128_to_lds(uint32_t ldsAddr, uint64_t gAddr) {
    asm volatile("global_load_async_to_lds_b128 %0, %1, off"
                 :: "v"(ldsAddr), "v"(gAddr) : "memory");
}

// ---------------------------------------------------------------------------
// CG matvec GEMM, bf16x3 (near-fp32): S = (Uh+Ul) x (Mh+Ml)^T, dropping Ul*Ml.
// Shapes fixed: M=NB(256), N=K=KBIG(1536).  tilesM=16, tilesN=24, 8 waves/blk
// -> all 8 waves of a block share one 64-row B tile, so B (hi+lo) is staged
// cooperatively into LDS with GLOBAL_LOAD_ASYNC_TO_LDS_B128, double-buffered
// on ASYNCcnt; compute reads fragments back via ds_load_b128.
// ---------------------------------------------------------------------------
__global__ __launch_bounds__(256) void gemm_cg_bf16x3_async(
    float* __restrict__ C,
    const __bf16* __restrict__ Ah, const __bf16* __restrict__ Al,
    const __bf16* __restrict__ Bh, const __bf16* __restrict__ Bl)
{
    const int lane   = threadIdx.x & 31;
    const int waveId = (blockIdx.x << 3) + (threadIdx.x >> 5);
    const int m0 = (waveId & 15) << 4;      // tilesM = 16
    const int p0 = (waveId >> 4) << 6;      // tilesN = 24
    const int khalf = lane >> 4;
    const int row   = lane & 15;

    __shared__ __align__(16) __bf16 smem[2][2][64][KCH];   // [buf][hi/lo][row][k] = 16KB

    // staging map: thread t moves one 16B segment per plane: row=t>>2, seg=t&3
    const int srow = threadIdx.x >> 2;
    const int sseg = threadIdx.x & 3;
    const uint64_t gH = (uint64_t)(uintptr_t)(Bh + (size_t)(p0 + srow) * KBIG + sseg * 8);
    const uint64_t gL = (uint64_t)(uintptr_t)(Bl + (size_t)(p0 + srow) * KBIG + sseg * 8);
    uint32_t lH[2], lL[2];
    lH[0] = (uint32_t)(uintptr_t)&smem[0][0][srow][sseg * 8];
    lL[0] = (uint32_t)(uintptr_t)&smem[0][1][srow][sseg * 8];
    lH[1] = (uint32_t)(uintptr_t)&smem[1][0][srow][sseg * 8];
    lL[1] = (uint32_t)(uintptr_t)&smem[1][1][srow][sseg * 8];

    // prologue: chunk 0 -> buffer 0
    async_b128_to_lds(lH[0], gH);
    async_b128_to_lds(lL[0], gL);

    v8f acc0 = {}, acc1 = {}, acc2 = {}, acc3 = {};
    const size_t aoff = (size_t)(m0 + row) * KBIG + 8 * khalf;
    const __bf16* aHp = Ah + aoff;
    const __bf16* aLp = Al + aoff;

    for (int c = 0; c < NCHUNK; ++c) {
        const int k   = c * KCH;
        const int buf = c & 1;
        if (c + 1 < NCHUNK) {
            const uint64_t go = (uint64_t)((c + 1) * KCH) * sizeof(__bf16);
            async_b128_to_lds(lH[buf ^ 1], gH + go);
            async_b128_to_lds(lL[buf ^ 1], gL + go);
            asm volatile("s_wait_asynccnt 0x2" ::: "memory");  // chunk c landed
        } else {
            asm volatile("s_wait_asynccnt 0x0" ::: "memory");
        }
        __syncthreads();                                        // LDS visible to all waves

        v16bf ah = load_frag_bf16(aHp + k);
        v16bf al = load_frag_bf16(aLp + k);

        v16bf b;
        b = load_frag_bf16(&smem[buf][0][row][8 * khalf]);
        acc0 = __builtin_amdgcn_wmma_f32_16x16x32_bf16(false, ah, false, b, (short)0, acc0, false, false);
        acc0 = __builtin_amdgcn_wmma_f32_16x16x32_bf16(false, al, false, b, (short)0, acc0, false, false);
        b = load_frag_bf16(&smem[buf][1][row][8 * khalf]);
        acc0 = __builtin_amdgcn_wmma_f32_16x16x32_bf16(false, ah, false, b, (short)0, acc0, false, false);

        b = load_frag_bf16(&smem[buf][0][16 + row][8 * khalf]);
        acc1 = __builtin_amdgcn_wmma_f32_16x16x32_bf16(false, ah, false, b, (short)0, acc1, false, false);
        acc1 = __builtin_amdgcn_wmma_f32_16x16x32_bf16(false, al, false, b, (short)0, acc1, false, false);
        b = load_frag_bf16(&smem[buf][1][16 + row][8 * khalf]);
        acc1 = __builtin_amdgcn_wmma_f32_16x16x32_bf16(false, ah, false, b, (short)0, acc1, false, false);

        b = load_frag_bf16(&smem[buf][0][32 + row][8 * khalf]);
        acc2 = __builtin_amdgcn_wmma_f32_16x16x32_bf16(false, ah, false, b, (short)0, acc2, false, false);
        acc2 = __builtin_amdgcn_wmma_f32_16x16x32_bf16(false, al, false, b, (short)0, acc2, false, false);
        b = load_frag_bf16(&smem[buf][1][32 + row][8 * khalf]);
        acc2 = __builtin_amdgcn_wmma_f32_16x16x32_bf16(false, ah, false, b, (short)0, acc2, false, false);

        b = load_frag_bf16(&smem[buf][0][48 + row][8 * khalf]);
        acc3 = __builtin_amdgcn_wmma_f32_16x16x32_bf16(false, ah, false, b, (short)0, acc3, false, false);
        acc3 = __builtin_amdgcn_wmma_f32_16x16x32_bf16(false, al, false, b, (short)0, acc3, false, false);
        b = load_frag_bf16(&smem[buf][1][48 + row][8 * khalf]);
        acc3 = __builtin_amdgcn_wmma_f32_16x16x32_bf16(false, ah, false, b, (short)0, acc3, false, false);

        __syncthreads();    // all waves done with buf before it is refilled
    }

#pragma unroll
    for (int v = 0; v < 8; ++v) {
        const int r = m0 + v + khalf * 8;
        const size_t cb = (size_t)r * KBIG + p0 + row;
        C[cb]      = acc0[v];
        C[cb + 16] = acc1[v];
        C[cb + 32] = acc2[v];
        C[cb + 48] = acc3[v];
    }
}

// ---------------------------------------------------------------------------
__device__ __forceinline__ float sigm(float v) { return 1.0f / (1.0f + __expf(-v)); }

__device__ __forceinline__ void bf16split(float v, __bf16& h, __bf16& l) {
    h = (__bf16)v;
    l = (__bf16)(v - (float)h);
}

// Gate nonlinearities + Jacobian coefficients.
__global__ __launch_bounds__(256) void lstm_gates(
    const float* __restrict__ Z, const float* __restrict__ cx,
    const float* __restrict__ b_ih, const float* __restrict__ b_hh,
    float* __restrict__ h_out, float* __restrict__ c_new,
    float* __restrict__ ci, float* __restrict__ cf, float* __restrict__ cgv)
{
    const int idx = blockIdx.x * 256 + threadIdx.x;     // < NB*HH
    const int h   = idx & (HH - 1);
    const size_t zb = (size_t)(idx >> 9) * (4 * HH);
    const float zi = Z[zb + h]          + b_ih[h]          + b_hh[h];
    const float zf = Z[zb + HH + h]     + b_ih[HH + h]     + b_hh[HH + h];
    const float zg = Z[zb + 2*HH + h]   + b_ih[2*HH + h]   + b_hh[2*HH + h];
    const float zo = Z[zb + 3*HH + h]   + b_ih[3*HH + h]   + b_hh[3*HH + h];
    const float i = sigm(zi), f = sigm(zf), g = tanhf(zg), o = sigm(zo);
    const float c  = cx[idx];
    const float cn = f * c + i * g;
    h_out[idx] = o * tanhf(cn);
    c_new[idx] = cn;
    ci[idx]  = i * (1.0f - i) * g;
    cf[idx]  = f * (1.0f - f) * c;
    cgv[idx] = (1.0f - g * g) * i;
}

// Split fp32 matrix into bf16 hi/lo pair (one-time, for M_big).
__global__ __launch_bounds__(256) void split_bf16(
    const float* __restrict__ src, __bf16* __restrict__ h, __bf16* __restrict__ l)
{
    const int idx = blockIdx.x * 256 + threadIdx.x;
    bf16split(src[idx], h[idx], l[idx]);
}

// CG init: x=0, r=d=c_new, rs=||r||^2, U(hi/lo) = cb .* d.  One block per n.
__global__ __launch_bounds__(256) void cg_init(
    const float* __restrict__ c_new, const float* __restrict__ ci,
    const float* __restrict__ cf, const float* __restrict__ cgv,
    float* __restrict__ x, float* __restrict__ r, float* __restrict__ d,
    __bf16* __restrict__ Uh, __bf16* __restrict__ Ul, float* __restrict__ rs)
{
    const int n = blockIdx.x, t = threadIdx.x;
    __shared__ float red[256];
    const size_t base = (size_t)n * HH, sb = (size_t)n * (3 * HH);
    float s = 0.0f;
#pragma unroll
    for (int jj = 0; jj < 2; ++jj) {
        const int j = t + jj * 256;
        const float v = c_new[base + j];
        x[base + j] = 0.0f; r[base + j] = v; d[base + j] = v;
        bf16split(ci[base + j]  * v, Uh[sb + j],        Ul[sb + j]);
        bf16split(cf[base + j]  * v, Uh[sb + HH + j],   Ul[sb + HH + j]);
        bf16split(cgv[base + j] * v, Uh[sb + 2*HH + j], Ul[sb + 2*HH + j]);
        s += v * v;
    }
    red[t] = s; __syncthreads();
    for (int st = 128; st > 0; st >>= 1) { if (t < st) red[t] += red[t + st]; __syncthreads(); }
    if (t == 0) rs[n] = red[0];
}

// One CG step per batch row n.
__global__ __launch_bounds__(256) void cg_step(
    float* __restrict__ x, float* __restrict__ r, float* __restrict__ d,
    __bf16* __restrict__ Uh, __bf16* __restrict__ Ul, const float* __restrict__ S,
    const float* __restrict__ ci, const float* __restrict__ cf,
    const float* __restrict__ cgv, float* __restrict__ rs)
{
    const int n = blockIdx.x, t = threadIdx.x;
    __shared__ float red[256];
    const size_t base = (size_t)n * HH, sb = (size_t)n * (3 * HH);
    const int j0 = t, j1 = t + 256;

    const float d0 = d[base + j0], d1 = d[base + j1];
    const float q0 = d0 + EPSV * (ci[base + j0] * S[sb + j0] +
                                  cf[base + j0] * S[sb + HH + j0] +
                                  cgv[base + j0] * S[sb + 2*HH + j0]);
    const float q1 = d1 + EPSV * (ci[base + j1] * S[sb + j1] +
                                  cf[base + j1] * S[sb + HH + j1] +
                                  cgv[base + j1] * S[sb + 2*HH + j1]);
    red[t] = d0 * q0 + d1 * q1; __syncthreads();
    for (int st = 128; st > 0; st >>= 1) { if (t < st) red[t] += red[t + st]; __syncthreads(); }
    const float dq = red[0]; __syncthreads();

    const float rsold = rs[n];
    const float alpha = (dq != 0.0f) ? rsold / dq : 0.0f;
    const float r0 = r[base + j0] - alpha * q0;
    const float r1 = r[base + j1] - alpha * q1;
    x[base + j0] += alpha * d0;  x[base + j1] += alpha * d1;
    r[base + j0] = r0;           r[base + j1] = r1;

    red[t] = r0 * r0 + r1 * r1; __syncthreads();
    for (int st = 128; st > 0; st >>= 1) { if (t < st) red[t] += red[t + st]; __syncthreads(); }
    const float rsnew = red[0];
    const float beta  = (rsold != 0.0f) ? rsnew / rsold : 0.0f;

    const float nd0 = r0 + beta * d0, nd1 = r1 + beta * d1;
    d[base + j0] = nd0; d[base + j1] = nd1;
    bf16split(ci[base + j0]  * nd0, Uh[sb + j0],        Ul[sb + j0]);
    bf16split(ci[base + j1]  * nd1, Uh[sb + j1],        Ul[sb + j1]);
    bf16split(cf[base + j0]  * nd0, Uh[sb + HH + j0],   Ul[sb + HH + j0]);
    bf16split(cf[base + j1]  * nd1, Uh[sb + HH + j1],   Ul[sb + HH + j1]);
    bf16split(cgv[base + j0] * nd0, Uh[sb + 2*HH + j0], Ul[sb + 2*HH + j0]);
    bf16split(cgv[base + j1] * nd1, Uh[sb + 2*HH + j1], Ul[sb + 2*HH + j1]);
    if (t == 0) rs[n] = rsnew;
}

__global__ __launch_bounds__(256) void copy_out(
    float* __restrict__ dst, const float* __restrict__ src)
{
    const int idx = blockIdx.x * 256 + threadIdx.x;
    dst[idx] = src[idx];
}

// ---------------------------------------------------------------------------
extern "C" void kernel_launch(void* const* d_in, const int* in_sizes, int n_in,
                              void* d_out, int out_size, void* d_ws, size_t ws_size,
                              hipStream_t stream) {
    (void)in_sizes; (void)n_in; (void)out_size; (void)ws_size;
    const float* x    = (const float*)d_in[0];
    const float* hx   = (const float*)d_in[1];
    const float* cx   = (const float*)d_in[2];
    const float* W_ih = (const float*)d_in[3];
    const float* W_hh = (const float*)d_in[4];
    const float* b_ih = (const float*)d_in[5];
    const float* b_hh = (const float*)d_in[6];
    float* out = (float*)d_out;              // [h_new (N*H) | prox_c (N*H)]
    float* ws  = (float*)d_ws;

    // fp32 workspace
    float* Z     = ws;                       // 256*2048
    float* c_new = Z + (size_t)NB * 4 * HH;
    float* ci    = c_new + (size_t)NB * HH;
    float* cf    = ci    + (size_t)NB * HH;
    float* cgv   = cf    + (size_t)NB * HH;
    float* Mb    = cgv   + (size_t)NB * HH;  // 1536*1536 fp32
    float* xs    = Mb  + (size_t)KBIG * KBIG;
    float* rv    = xs  + (size_t)NB * HH;
    float* dv    = rv  + (size_t)NB * HH;
    float* S     = dv  + (size_t)NB * HH;    // 256*1536
    float* rs    = S   + (size_t)NB * KBIG;
    // bf16 workspace (starts 16B-aligned: float count above is divisible by 4)
    __bf16* Mh = (__bf16*)(rs + 256);
    __bf16* Ml = Mh + (size_t)KBIG * KBIG;
    __bf16* Uh = Ml + (size_t)KBIG * KBIG;
    __bf16* Ul = Uh + (size_t)NB * KBIG;

    // --- z = x*W_ih^T + hx*W_hh^T  (exact f32 WMMA; biases folded in gates) ---
    gemm_tn_wmma<<<64, 256, 0, stream>>>(Z, x,  W_ih, NB, 4*HH, DD, DD, DD, 4*HH, 0);
    gemm_tn_wmma<<<64, 256, 0, stream>>>(Z, hx, W_hh, NB, 4*HH, HH, HH, HH, 4*HH, 1);

    // --- weight Gram M_big = W_ifg * W_ifg^T (exact f32 WMMA, one-time) ---
    gemm_tn_wmma<<<288, 256, 0, stream>>>(Mb, W_ih, W_ih, KBIG, KBIG, DD, DD, DD, KBIG, 0);
    split_bf16<<<(KBIG * KBIG) / 256, 256, 0, stream>>>(Mb, Mh, Ml);

    // --- gates, h_new, c_new, Jacobian coefficients ---
    lstm_gates<<<(NB * HH) / 256, 256, 0, stream>>>(Z, cx, b_ih, b_hh,
                                                    out, c_new, ci, cf, cgv);

    // --- batched CG on (I + eps * diag(ca) M_ab diag(cb)) p = c_new ---
    cg_init<<<NB, 256, 0, stream>>>(c_new, ci, cf, cgv, xs, rv, dv, Uh, Ul, rs);
    for (int it = 0; it < CG_ITERS; ++it) {
        // S = U * M_big on the bf16 matrix pipe, B tile async-staged in LDS
        gemm_cg_bf16x3_async<<<48, 256, 0, stream>>>(S, Uh, Ul, Mh, Ml);
        cg_step<<<NB, 256, 0, stream>>>(xs, rv, dv, Uh, Ul, S, ci, cf, cgv, rs);
    }

    // --- prox_c output ---
    copy_out<<<(NB * HH) / 256, 256, 0, stream>>>(out + (size_t)NB * HH, xs);
}
